// GCNFlowArchEmbedder_40785009443628
// MI455X (gfx1250) — matrix operand order repaired and checked by reading, not compiled, not run on hardware
//
#include <hip/hip_runtime.h>
#include <hip/hip_bf16.h>
#include <stdint.h>

typedef __attribute__((ext_vector_type(16))) _Float16 v16h;
typedef __attribute__((ext_vector_type(8)))  float    v8f;

// Workspace layout (floats):
//   [0,1024)     sig0[8][128]
//   [1024,2048)  sig1[8][128]
//   [2048,2560)  s0_init[2cg][2r][128]
//   [2560,2816)  s0_zero[2cg][128]
//   [2816,3328)  s1_init[2cg][2r][128]
//   [3328, +16384 halves) w1T f16, N-major: w1T[n*128+k] = (f16)w1[k][n]
#define WS_CONST_FLOATS 3328

__global__ __launch_bounds__(128)
void gcn_precompute(const float* __restrict__ ine,   // init_node_emb [2][2][48]
                    const float* __restrict__ one,   // other_node_emb [2][1][48]
                    const float* __restrict__ ope,   // op_embs [8][48]
                    const float* __restrict__ xhw,   // [48][96]
                    const float* __restrict__ xhb,   // [96]
                    const float* __restrict__ w0,    // [96][128]
                    const float* __restrict__ a0w,   // [48][128]
                    const float* __restrict__ a0b,   // [128]
                    const float* __restrict__ w1,    // [128][128]
                    const float* __restrict__ a1w,   // [48][128]
                    const float* __restrict__ a1b,   // [128]
                    float* __restrict__ ws)
{
    __shared__ float xs[6][96];    // x rows: 4 init (cg,r) + 2 "other" (cg)
    __shared__ float y0i[4][128];  // relu(support0) rows 0,1 per cg
    const int t = threadIdx.x;
    _Float16* w1T = (_Float16*)(ws + WS_CONST_FLOATS);

    if (t < 128) {
        for (int o = 0; o < 8; ++o) {
            float a0 = a0b[t], a1 = a1b[t];
            for (int k = 0; k < 48; ++k) {
                float e = ope[o*48 + k];
                a0 += e * a0w[k*128 + t];
                a1 += e * a1w[k*128 + t];
            }
            ws[o*128 + t]        = 1.f / (1.f + __expf(-a0));
            ws[1024 + o*128 + t] = 1.f / (1.f + __expf(-a1));
        }
    }
    if (t < 96) {
        for (int cr = 0; cr < 4; ++cr) {
            float a = xhb[t];
            for (int k = 0; k < 48; ++k) a += ine[cr*48 + k] * xhw[k*96 + t];
            xs[cr][t] = a;
        }
        for (int cg = 0; cg < 2; ++cg) {
            float a = xhb[t];
            for (int k = 0; k < 48; ++k) a += one[cg*48 + k] * xhw[k*96 + t];
            xs[4 + cg][t] = a;
        }
    }
    __syncthreads();
    if (t < 128) {
        for (int cr = 0; cr < 4; ++cr) {
            float a = 0.f;
            for (int j = 0; j < 96; ++j) a += xs[cr][j] * w0[j*128 + t];
            ws[2048 + cr*128 + t] = a;
            y0i[cr][t] = fmaxf(a, 0.f);
        }
        for (int cg = 0; cg < 2; ++cg) {
            float z = 0.f;
            for (int j = 0; j < 96; ++j) z += xs[4 + cg][j] * w0[j*128 + t];
            ws[2560 + cg*128 + t] = z;
        }
    }
    __syncthreads();
    if (t < 128) {
        for (int cr = 0; cr < 4; ++cr) {
            float a = 0.f;
            for (int j = 0; j < 128; ++j) a += y0i[cr][j] * w1[j*128 + t];
            ws[2816 + cr*128 + t] = a;
        }
    }
    for (int idx = t; idx < 128*128; idx += 128) {
        int n = idx >> 7, k = idx & 127;
        w1T[n*128 + k] = (_Float16)w1[k*128 + n];
    }
}

// 8 (b,cg) pairs per block; 32 y0 rows -> 2 WMMA M-tiles; N=128 -> 8 N-tiles.
__global__ __launch_bounds__(256)
void gcn_main(const int* __restrict__ archs,
              const float* __restrict__ ws,
              float* __restrict__ out)
{
    __shared__ float    cbuf[WS_CONST_FLOATS];
    __shared__ int      eF[8][8], eO[8][8];
    __shared__ _Float16 y0s[32 * 136];   // padded stride (conflict-free A-frag loads)
    __shared__ float    s1s[32 * 132];   // padded stride

    const int tid = threadIdx.x;
    const int P0  = blockIdx.x * 8;      // global (b,cg) pair base; P = b*2+cg

    for (int i = tid; i < WS_CONST_FLOATS; i += 256) cbuf[i] = ws[i];
    if (tid < 128) {
        int p = tid >> 4, s = (tid >> 3) & 1, e = tid & 7;
        int v = archs[(P0 + p) * 16 + s * 8 + e];
        if (s == 0) eF[p][e] = v; else eO[p][e] = v;
    }
    __syncthreads();

    // ---- Phase 1: layer-0 aggregate -> y0 rows (nodes 2..5), f16 into LDS ----
    {
        const float* sig0 = cbuf;
        const float* s0i  = cbuf + 2048;
        const float* s0z  = cbuf + 2560;
        for (int idx = tid; idx < 4096; idx += 256) {
            int p = idx >> 9, i = (idx >> 7) & 3, d = idx & 127;
            int cg = (P0 + p) & 1;
            const float* s0zc = s0z + cg * 128;
            float acc = s0zc[d];
            int f0 = eF[p][2*i],   o0 = eO[p][2*i];
            int f1 = eF[p][2*i+1], o1 = eO[p][2*i+1];
            float Sa = (f0 < 2) ? s0i[(cg*2 + f0)*128 + d] : s0zc[d];
            float Sb = (f1 < 2) ? s0i[(cg*2 + f1)*128 + d] : s0zc[d];
            if (f0 == f1) acc += 2.f * sig0[o1*128 + d] * Sb;     // adj=2, last op wins
            else          acc += sig0[o0*128 + d] * Sa + sig0[o1*128 + d] * Sb;
            y0s[(p*4 + i) * 136 + d] = (_Float16)fmaxf(acc, 0.f);
        }
    }
    __syncthreads();

    // ---- Phase 2: support1 = y0 @ w1 via v_wmma_f32_16x16x32_f16 ----
    {
        const int w = tid >> 5, lane = tid & 31;
        const int mtile = w >> 2, n0 = (w & 3) * 32;
        const int hiK = (lane & 16) ? 1 : 0;
        const uint32_t* w1T32 = (const uint32_t*)(ws + WS_CONST_FLOATS);
        const uint32_t* y32   = (const uint32_t*)y0s;

        union HV { uint32_t u[8]; v16h h; };
        HV B[2][4];
        #pragma unroll
        for (int nt = 0; nt < 2; ++nt) {
            int n = n0 + nt*16 + (lane & 15);
            #pragma unroll
            for (int k = 0; k < 4; ++k)
                #pragma unroll
                for (int v = 0; v < 8; ++v)   // B: lane=N; K pairs packed, lanes>=16 hold K+16
                    B[nt][k].u[v] = w1T32[n*64 + k*16 + hiK*8 + v];
        }
        v8f c0 = {}; v8f c1 = {};
        const int r = lane & 15;
        #pragma unroll
        for (int k = 0; k < 4; ++k) {
            HV a;
            int b32 = (((mtile*16 + r) * 136) + k*32 + hiK*16) >> 1;
            #pragma unroll
            for (int v = 0; v < 8; ++v)        // A: K pairs packed, lanes>=16 hold K+16
                a.u[v] = y32[b32 + v];
            c0 = __builtin_amdgcn_wmma_f32_16x16x32_f16(false, a.h, false, B[0][k].h,
                                                        (short)0, c0, false, false);
            c1 = __builtin_amdgcn_wmma_f32_16x16x32_f16(false, a.h, false, B[1][k].h,
                                                        (short)0, c1, false, false);
        }
        const int m0 = mtile*16 + hiK*8;
        const int nc = n0 + (lane & 15);
        #pragma unroll
        for (int rr = 0; rr < 8; ++rr) {
            s1s[(m0 + rr) * 132 + nc]      = c0[rr];
            s1s[(m0 + rr) * 132 + nc + 16] = c1[rr];
        }
    }
    __syncthreads();

    // ---- Phase 3: layer-1 aggregate + mean over nodes 2..5, write output ----
    {
        const float* sig1 = cbuf + 1024;
        const float* s1i  = cbuf + 2816;
        for (int idx = tid; idx < 1024; idx += 256) {
            int p = idx >> 7, d = idx & 127;
            int P = P0 + p, cg = P & 1;
            float sum = 0.f;
            #pragma unroll
            for (int i = 0; i < 4; ++i) {
                sum += s1s[(p*4 + i) * 132 + d];
                int f0 = eF[p][2*i],   o0 = eO[p][2*i];
                int f1 = eF[p][2*i+1], o1 = eO[p][2*i+1];
                float Sa = (f0 < 2) ? s1i[(cg*2 + f0)*128 + d] : s1s[(p*4 + f0 - 2)*132 + d];
                float Sb = (f1 < 2) ? s1i[(cg*2 + f1)*128 + d] : s1s[(p*4 + f1 - 2)*132 + d];
                if (f0 == f1) sum += 2.f * sig1[o1*128 + d] * Sb;
                else          sum += sig1[o0*128 + d] * Sa + sig1[o1*128 + d] * Sb;
            }
            out[P * 128 + d] = 0.25f * sum;
        }
    }
}

extern "C" void kernel_launch(void* const* d_in, const int* in_sizes, int n_in,
                              void* d_out, int out_size, void* d_ws, size_t ws_size,
                              hipStream_t stream) {
    const int*   archs = (const int*)  d_in[0];
    const float* ine   = (const float*)d_in[1];
    const float* one   = (const float*)d_in[2];
    const float* ope   = (const float*)d_in[3];
    const float* xhw   = (const float*)d_in[4];
    const float* xhb   = (const float*)d_in[5];
    const float* w0    = (const float*)d_in[6];
    const float* a0w   = (const float*)d_in[7];
    const float* a0b   = (const float*)d_in[8];
    const float* w1    = (const float*)d_in[9];
    const float* a1w   = (const float*)d_in[10];
    const float* a1b   = (const float*)d_in[11];
    float* ws = (float*)d_ws;

    gcn_precompute<<<1, 128, 0, stream>>>(ine, one, ope, xhw, xhb,
                                          w0, a0w, a0b, w1, a1w, a1b, ws);
    // 8192 batch * 2 cg / 8 pairs-per-block = 2048 blocks
    gcn_main<<<2048, 256, 0, stream>>>(archs, ws, (float*)d_out);
}
